// Block_49443663511610
// MI455X (gfx1250) — matrix operand order repaired and checked by reading, not compiled, not run on hardware
//
#include <hip/hip_runtime.h>

#define DEV __device__ __forceinline__

typedef __attribute__((ext_vector_type(16))) _Float16 v16h;
typedef __attribute__((ext_vector_type(8)))  _Float16 v8h;
typedef __attribute__((ext_vector_type(8)))  float    v8f;

static constexpr int B_  = 256;   // batch
static constexpr int T_  = 256;   // seq
static constexpr int D_  = 256;   // n_embd
static constexpr int H_  = 8;     // heads
static constexpr int HS_ = 32;    // head size
static constexpr int FF_ = 1024;  // ffn hidden
static constexpr int TOK = B_ * T_;

// ---------------------------------------------------------------------------
// WMMA tile loaders (wave32): A and B share contiguous k order, not the ISA table order.
// A (16x32 f16): lane holds row (lane&15); K = (lane>>4)*16 .. +15 (matches B).
// B (32x16 f16): lane holds COLUMN (lane&15); K = (lane>>4)*16 .. +15.
//   -> load B from the row-major transpose ("[N][K]" layout): 16 contiguous f16.
// C/D (16x16 f32): lane l, vgpr r -> (M = r + 8*(l>>4), N = l&15).
// ---------------------------------------------------------------------------
DEV v16h load_tile_a(const _Float16* base, int ld) {
  const int lane = threadIdx.x & 31;
  const _Float16* r = base + (lane & 15) * ld + ((lane >> 4) << 4);
  v8h lo = *(const v8h*)(r);
  v8h hi = *(const v8h*)(r + 8);
  v16h a;
#pragma unroll
  for (int i = 0; i < 8; ++i) { a[i] = lo[i]; a[i + 8] = hi[i]; }
  return a;
}

DEV v16h load_tile_b(const _Float16* base, int ld) {
  const int lane = threadIdx.x & 31;
  const _Float16* r = base + (lane & 15) * ld + ((lane >> 4) << 4);
  return *(const v16h*)(r);
}

DEV v8f wmma_f16(v16h a, v16h b, v8f c) {
  return __builtin_amdgcn_wmma_f32_16x16x32_f16(false, a, false, b, (short)0, c,
                                                false, false);
}

DEV float wave_sum(float v) {
#pragma unroll
  for (int o = 16; o > 0; o >>= 1) v += __shfl_xor(v, o, 32);
  return v;
}
DEV float wave_max(float v) {
#pragma unroll
  for (int o = 16; o > 0; o >>= 1) v = fmaxf(v, __shfl_xor(v, o, 32));
  return v;
}

// ---------------------------------------------------------------------------
// Weight prep: fp32 -> f16, transposed to [out][in] so B-operands are
// contiguous 32B loads per lane.
// ---------------------------------------------------------------------------
__global__ void k_prep(const float* __restrict__ Wq, const float* __restrict__ Wk,
                       const float* __restrict__ Wv, const float* __restrict__ Wo,
                       const float* __restrict__ W1, const float* __restrict__ W2,
                       _Float16* __restrict__ wqt, _Float16* __restrict__ wkt,
                       _Float16* __restrict__ wvt, _Float16* __restrict__ wot,
                       _Float16* __restrict__ w1t, _Float16* __restrict__ w2t) {
  int i = blockIdx.x * 256 + threadIdx.x;
  if (i < 4 * 65536) {                       // four 256x256 transposes
    int seg = i >> 16, j = i & 65535;
    int o = j >> 8, d = j & 255;
    const float*    W  = seg == 0 ? Wq : seg == 1 ? Wk : seg == 2 ? Wv : Wo;
    _Float16*       Tp = seg == 0 ? wqt : seg == 1 ? wkt : seg == 2 ? wvt : wot;
    Tp[o * 256 + d] = (_Float16)W[d * 256 + o];   // Wq is [d][h*32+f]
  } else {
    int j = i - 4 * 65536;
    if (j < 262144) {                        // W1 [256][1024] -> w1t [1024][256]
      int o = j >> 8, d = j & 255;
      w1t[o * 256 + d] = (_Float16)W1[d * 1024 + o];
    } else {                                 // W2 [1024][256] -> w2t [256][1024]
      j -= 262144;
      int d = j >> 10, o = j & 1023;
      w2t[d * 1024 + o] = (_Float16)W2[o * 256 + d];
    }
  }
}

// ---------------------------------------------------------------------------
// LayerNorm 1: wave per token, fp32 stats, f16 output.
// ---------------------------------------------------------------------------
__global__ void k_ln1(const float* __restrict__ x, const float* __restrict__ g,
                      const float* __restrict__ b, _Float16* __restrict__ h) {
  const int token = blockIdx.x * 8 + (threadIdx.x >> 5);
  const int lane  = threadIdx.x & 31;
  const float* xr = x + (size_t)token * D_;
  float s = 0.f, s2 = 0.f;
  for (int c = lane; c < D_; c += 32) { float v = xr[c]; s += v; s2 += v * v; }
  s  = wave_sum(s);
  s2 = wave_sum(s2);
  const float mu = s * (1.f / D_);
  const float rs = rsqrtf(s2 * (1.f / D_) - mu * mu + 1e-5f);
  _Float16* hr = h + (size_t)token * D_;
  for (int c = lane; c < D_; c += 32)
    hr[c] = (_Float16)((xr[c] - mu) * rs * g[c] + b[c]);
}

// ---------------------------------------------------------------------------
// QKV projection: wave computes 16 tokens x 32 outs (one head) for one of
// q/k/v. q,k stored [B,H,T,32]; v stored transposed [B,H,32,T] so attention's
// wei@V B-operand is contiguous.
// ---------------------------------------------------------------------------
__global__ void k_qkv(const _Float16* __restrict__ h,
                      const _Float16* __restrict__ wqt,
                      const _Float16* __restrict__ wkt,
                      const _Float16* __restrict__ wvt,
                      _Float16* __restrict__ qb, _Float16* __restrict__ kb,
                      _Float16* __restrict__ vtb) {
  const int wid   = blockIdx.x;
  const int which = wid % 3;
  const int head  = (wid / 3) & 7;
  const int tt    = wid / 24;
  const _Float16* wt = (which == 0) ? wqt : (which == 1) ? wkt : wvt;
  const int o0 = head * HS_;
  const _Float16* hrow = h + (size_t)tt * 16 * D_;
  v8f c0 = {}, c1 = {};
#pragma unroll
  for (int kk = 0; kk < 8; ++kk) {
    v16h a  = load_tile_a(hrow + kk * 32, D_);
    v16h b0 = load_tile_b(wt + (size_t)o0 * D_ + kk * 32, D_);
    v16h b1 = load_tile_b(wt + (size_t)(o0 + 16) * D_ + kk * 32, D_);
    c0 = wmma_f16(a, b0, c0);
    c1 = wmma_f16(a, b1, c1);
  }
  const int lane = threadIdx.x & 31;
  const int n = lane & 15, moff = (lane >> 4) * 8;
  const int tok0 = tt * 16;
  const int bidx = tok0 >> 8;       // T_==256: tile never crosses batch row
  const int tloc = tok0 & 255;
  if (which < 2) {
    _Float16* dst =
        (which == 0 ? qb : kb) + ((size_t)(bidx * H_ + head) * T_ + tloc) * HS_;
#pragma unroll
    for (int r = 0; r < 8; ++r) {
      dst[(moff + r) * HS_ + n]      = (_Float16)c0[r];
      dst[(moff + r) * HS_ + 16 + n] = (_Float16)c1[r];
    }
  } else {
    _Float16* dst = vtb + (size_t)(bidx * H_ + head) * HS_ * T_ + tloc;
#pragma unroll
    for (int r = 0; r < 8; ++r) {
      dst[(size_t)n * T_ + moff + r]        = (_Float16)c0[r];
      dst[(size_t)(16 + n) * T_ + moff + r] = (_Float16)c1[r];
    }
  }
}

// ---------------------------------------------------------------------------
// Causal attention: wave per (b, h, 16-query block). K=32==head_size so each
// 16x16 score tile is ONE wmma. fp32 softmax in LDS, f16 weights, wei@V via
// wmma against V^T.
// ---------------------------------------------------------------------------
__global__ void k_attn(const _Float16* __restrict__ qb,
                       const _Float16* __restrict__ kb,
                       const _Float16* __restrict__ vtb,
                       _Float16* __restrict__ attn) {
  __shared__ __attribute__((aligned(32))) float    sc[16][T_];
  __shared__ __attribute__((aligned(32))) _Float16 wf[16][T_];
  const int wid = blockIdx.x;
  const int tb = wid & 15;
  const int hh = (wid >> 4) & 7;
  const int bb = wid >> 7;
  const int t0 = tb * 16;
  const int lane = threadIdx.x & 31;
  const size_t bh = (size_t)(bb * H_ + hh);
  const _Float16* qp = qb + bh * T_ * HS_;
  const _Float16* kp = kb + bh * T_ * HS_;
  const _Float16* vt = vtb + bh * HS_ * T_;

  v16h aq = load_tile_a(qp + t0 * HS_, HS_);
  const float scale = 0.1767766952966369f;  // 1/sqrt(32)
  for (int sb = 0; sb <= tb; ++sb) {
    v16h bk = load_tile_b(kp + sb * 16 * HS_, HS_);  // B = K^T via K rows
    v8f c = {};
    c = wmma_f16(aq, bk, c);
    const int n = lane & 15, moff = (lane >> 4) * 8;
#pragma unroll
    for (int r = 0; r < 8; ++r) sc[moff + r][sb * 16 + n] = c[r] * scale;
  }
  __syncthreads();
  const int len = t0 + 16;
  const int lp  = (len + 31) & ~31;  // pad to K-chunk of 32
  for (int r = 0; r < 16; ++r) {
    const int rl = t0 + r + 1;       // causal row length
    float mx = -3.0e38f;
    for (int s = lane; s < rl; s += 32) mx = fmaxf(mx, sc[r][s]);
    mx = wave_max(mx);
    float sum = 0.f;
    for (int s = lane; s < rl; s += 32) {
      float e = __expf(sc[r][s] - mx);
      sum += e;
      sc[r][s] = e;
    }
    sum = wave_sum(sum);
    const float inv = 1.f / sum;
    for (int s = lane; s < lp; s += 32)
      wf[r][s] = (s < rl) ? (_Float16)(sc[r][s] * inv) : (_Float16)0.f;
  }
  __syncthreads();
  v8f a0 = {}, a1 = {};
  for (int kk = 0; kk < (lp >> 5); ++kk) {
    v16h aw = load_tile_a(&wf[0][kk * 32], T_);
    v16h b0 = load_tile_b(vt + 0 * T_  + kk * 32, T_);   // rows f=0..15 of V^T
    v16h b1 = load_tile_b(vt + 16 * T_ + kk * 32, T_);   // rows f=16..31
    a0 = wmma_f16(aw, b0, a0);
    a1 = wmma_f16(aw, b1, a1);
  }
  _Float16* dst = attn + ((size_t)bb * T_ + t0) * D_ + hh * HS_;
  const int n = lane & 15, moff = (lane >> 4) * 8;
#pragma unroll
  for (int r = 0; r < 8; ++r) {
    dst[(moff + r) * D_ + n]      = (_Float16)a0[r];
    dst[(moff + r) * D_ + 16 + n] = (_Float16)a1[r];
  }
}

// ---------------------------------------------------------------------------
// Output projection + residual: x1 = x + attn @ Wo + bo (fp32 out).
// ---------------------------------------------------------------------------
__global__ void k_oproj(const _Float16* __restrict__ attn,
                        const _Float16* __restrict__ wot,
                        const float* __restrict__ x, const float* __restrict__ bo,
                        float* __restrict__ x1) {
  const int wid = blockIdx.x;
  const int nb = wid & 7;
  const int tt = wid >> 3;
  const int o0 = nb * 32;
  const _Float16* arow = attn + (size_t)tt * 16 * D_;
  v8f c0 = {}, c1 = {};
#pragma unroll
  for (int kk = 0; kk < 8; ++kk) {
    v16h a  = load_tile_a(arow + kk * 32, D_);
    v16h b0 = load_tile_b(wot + (size_t)o0 * D_ + kk * 32, D_);
    v16h b1 = load_tile_b(wot + (size_t)(o0 + 16) * D_ + kk * 32, D_);
    c0 = wmma_f16(a, b0, c0);
    c1 = wmma_f16(a, b1, c1);
  }
  const int lane = threadIdx.x & 31;
  const int n = lane & 15, moff = (lane >> 4) * 8;
#pragma unroll
  for (int r = 0; r < 8; ++r) {
    const size_t tok = (size_t)tt * 16 + moff + r;
    const int d0 = o0 + n, d1 = o0 + 16 + n;
    x1[tok * D_ + d0] = x[tok * D_ + d0] + bo[d0] + c0[r];
    x1[tok * D_ + d1] = x[tok * D_ + d1] + bo[d1] + c1[r];
  }
}

// ---------------------------------------------------------------------------
// Fused LN2 + FFN + residual: wave per 16-token tile. h2 and the 16x1024
// hidden activation live in LDS (40KB). GEMM2 runs K-outer with 16
// accumulator tiles so each LDS A-tile is read exactly once.
// ---------------------------------------------------------------------------
__global__ void k_ffn(const float* __restrict__ x1,
                      const float* __restrict__ lng, const float* __restrict__ lnb,
                      const _Float16* __restrict__ w1t,
                      const float* __restrict__ bias1,
                      const _Float16* __restrict__ w2t,
                      const float* __restrict__ bias2,
                      float* __restrict__ out) {
  __shared__ __attribute__((aligned(32))) _Float16 h2[16][D_];
  __shared__ __attribute__((aligned(32))) _Float16 f1[16][FF_];
  const int tt = blockIdx.x;
  const int lane = threadIdx.x & 31;
  const float* xr = x1 + (size_t)tt * 16 * D_;
  for (int r = 0; r < 16; ++r) {
    float s = 0.f, s2 = 0.f;
    for (int c = lane; c < D_; c += 32) { float v = xr[r * D_ + c]; s += v; s2 += v * v; }
    s  = wave_sum(s);
    s2 = wave_sum(s2);
    const float mu = s * (1.f / D_);
    const float rs = rsqrtf(s2 * (1.f / D_) - mu * mu + 1e-5f);
    for (int c = lane; c < D_; c += 32)
      h2[r][c] = (_Float16)((xr[r * D_ + c] - mu) * rs * lng[c] + lnb[c]);
  }
  __syncthreads();
  const int n = lane & 15, moff = (lane >> 4) * 8;

  // GEMM1: h2 (16x256) @ W1^T -> f1 (16x1024), ReLU.  A-tiles hoisted.
  v16h ah[8];
#pragma unroll
  for (int kk = 0; kk < 8; ++kk) ah[kk] = load_tile_a(&h2[0][kk * 32], D_);
  for (int n1 = 0; n1 < FF_ / 16; ++n1) {
    if (n1 + 1 < FF_ / 16)   // hint next B row-group toward L2 (global_prefetch)
      __builtin_prefetch(w1t + (size_t)((n1 + 1) * 16) * D_ + (lane & 15) * D_, 0, 3);
    v8f c = {};
#pragma unroll
    for (int kk = 0; kk < 8; ++kk)
      c = wmma_f16(ah[kk], load_tile_b(w1t + (size_t)(n1 * 16) * D_ + kk * 32, D_), c);
    const float bb = bias1[n1 * 16 + n];
#pragma unroll
    for (int r = 0; r < 8; ++r)
      f1[moff + r][n1 * 16 + n] = (_Float16)fmaxf(c[r] + bb, 0.f);
  }
  __syncthreads();

  // GEMM2: f1 (16x1024) @ W2^T -> 16x256, K-outer, 16 accumulators.
  v8f acc[16];
#pragma unroll
  for (int i = 0; i < 16; ++i) acc[i] = (v8f){};
  for (int kk = 0; kk < FF_ / 32; ++kk) {
    v16h a = load_tile_a(&f1[0][kk * 32], FF_);
#pragma unroll
    for (int n2 = 0; n2 < 16; ++n2) {
      v16h b = load_tile_b(w2t + (size_t)(n2 * 16) * FF_ + kk * 32, FF_);
      acc[n2] = wmma_f16(a, b, acc[n2]);
    }
  }
#pragma unroll
  for (int n2 = 0; n2 < 16; ++n2) {
    const float bb = bias2[n2 * 16 + n];
#pragma unroll
    for (int r = 0; r < 8; ++r) {
      const size_t tok = (size_t)tt * 16 + moff + r;
      out[tok * D_ + n2 * 16 + n] = xr[(moff + r) * D_ + n2 * 16 + n] + bb + acc[n2][r];
    }
  }
}

// ---------------------------------------------------------------------------
extern "C" void kernel_launch(void* const* d_in, const int* in_sizes, int n_in,
                              void* d_out, int out_size, void* d_ws, size_t ws_size,
                              hipStream_t stream) {
  (void)in_sizes; (void)n_in; (void)out_size; (void)ws_size;
  const float* x   = (const float*)d_in[0];
  const float* Wq  = (const float*)d_in[1];
  const float* Wk  = (const float*)d_in[2];
  const float* Wv  = (const float*)d_in[3];
  const float* Wo  = (const float*)d_in[4];
  const float* bo  = (const float*)d_in[5];
  const float* g1  = (const float*)d_in[6];
  const float* be1 = (const float*)d_in[7];
  const float* g2  = (const float*)d_in[8];
  const float* be2 = (const float*)d_in[9];
  const float* W1  = (const float*)d_in[10];
  const float* b1  = (const float*)d_in[11];
  const float* W2  = (const float*)d_in[12];
  const float* b2  = (const float*)d_in[13];
  float* out = (float*)d_out;

  // Workspace carve-up (~203 MB total; 256B aligned slices).
  char* w = (char*)d_ws;
  size_t off = 0;
  auto take = [&](size_t bytes) -> char* {
    char* p = w + off;
    off += (bytes + 255) & ~(size_t)255;
    return p;
  };
  _Float16* wqt = (_Float16*)take(65536 * 2);
  _Float16* wkt = (_Float16*)take(65536 * 2);
  _Float16* wvt = (_Float16*)take(65536 * 2);
  _Float16* wot = (_Float16*)take(65536 * 2);
  _Float16* w1t = (_Float16*)take(262144 * 2);
  _Float16* w2t = (_Float16*)take(262144 * 2);
  _Float16* hb  = (_Float16*)take((size_t)TOK * D_ * 2);
  _Float16* qb  = (_Float16*)take((size_t)TOK * D_ * 2);
  _Float16* kb  = (_Float16*)take((size_t)TOK * D_ * 2);
  _Float16* vtb = (_Float16*)take((size_t)TOK * D_ * 2);
  float*    x1  = (float*)   take((size_t)TOK * D_ * 4);
  _Float16* attn = hb;  // h consumed by k_qkv before k_attn writes -> safe reuse

  k_prep <<<3072, 256, 0, stream>>>(Wq, Wk, Wv, Wo, W1, W2,
                                    wqt, wkt, wvt, wot, w1t, w2t);
  k_ln1  <<<TOK / 8, 256, 0, stream>>>(x, g1, be1, hb);
  k_qkv  <<<(TOK / 16) * H_ * 3, 32, 0, stream>>>(hb, wqt, wkt, wvt, qb, kb, vtb);
  k_attn <<<B_ * H_ * (T_ / 16), 32, 0, stream>>>(qb, kb, vtb, attn);
  k_oproj<<<(TOK / 16) * (D_ / 32), 32, 0, stream>>>(attn, wot, x, bo, x1);
  k_ffn  <<<TOK / 16, 32, 0, stream>>>(x1, g2, be2, w1t, b1, w2t, b2, out);
}
